// CosformerAttention_9131100471586
// MI455X (gfx1250) — matrix operand order repaired and checked by reading, not compiled, not run on hardware
//
#include <hip/hip_runtime.h>

typedef __attribute__((ext_vector_type(16))) __bf16 v16bf;
typedef __attribute__((ext_vector_type(8)))  __bf16 v8bf;
typedef __attribute__((ext_vector_type(8)))  float  v8f;

#define D_MODEL 2048
#define SEQ     4096
#define NHEADS  16
#define DKH     128
#define NBATCH  4
#define MTOT    (NBATCH * SEQ)      // 16384
#define NZ      (NBATCH * NHEADS)   // 64

// LDS staging for the A tile: 128 rows x 32 k (bf16) = 64 B/row,
// padded to 80 B pitch (16B aligned; 20-bank stride = 5 16B groups, 5 coprime with 16 groups of 64 banks
// -> conflict-free ds_load_b128 across the 16 lanes of a fragment).
#define LDS_PITCH 80
#define LDS_BUF   (128 * LDS_PITCH)   // 10240 B per buffer, x2 buffers

union FragBF {
    v16bf v;
    v8bf  h[2];
};

enum { MODE_Q = 0, MODE_KT, MODE_VT, MODE_KVT, MODE_ATTN, MODE_OUT };

// Compiler-only scheduling fence: pin all issued memory ops before the MMA
// sequence so the scheduler staggers the dependency waits instead of
// serializing load->wait0->mma per fragment.
#define SCHED_FENCE() asm volatile("" ::: "memory")

// ---------------------------------------------------------------------------
// f32 -> bf16 conversion (4 elements / thread)
// ---------------------------------------------------------------------------
__global__ __launch_bounds__(256)
void cvt_f32_to_bf16(const float* __restrict__ in, __bf16* __restrict__ out) {
    size_t i = ((size_t)blockIdx.x * blockDim.x + threadIdx.x) * 4;
    float4 v = *(const float4*)(in + i);
    out[i + 0] = (__bf16)v.x;
    out[i + 1] = (__bf16)v.y;
    out[i + 2] = (__bf16)v.z;
    out[i + 3] = (__bf16)v.w;
}

// ---------------------------------------------------------------------------
// Ksum[z][d] = sum_n KT[z][d][n]   (KT rows are contiguous length SEQ)
// ---------------------------------------------------------------------------
__global__ __launch_bounds__(256)
void ksum_kernel(const __bf16* __restrict__ KT, float* __restrict__ Ksum) {
    int wave = threadIdx.x >> 5;
    int lane = threadIdx.x & 31;
    int row  = blockIdx.x * 8 + wave;       // 0 .. NZ*DKH-1
    const __bf16* p = KT + (size_t)row * SEQ;
    float s = 0.f;
    for (int i = lane; i < SEQ; i += 32) s += (float)p[i];
    #pragma unroll
    for (int off = 16; off > 0; off >>= 1) s += __shfl_xor(s, off, 32);
    if (lane == 0) Ksum[row] = s;
}

// ---------------------------------------------------------------------------
// Zden[z][n] = 1 / (dot(Q[z][n][:], Ksum[z][:]) + eps)
// ---------------------------------------------------------------------------
__global__ __launch_bounds__(256)
void zden_kernel(const __bf16* __restrict__ Qb, const float* __restrict__ Ksum,
                 float* __restrict__ Zden) {
    int wave = threadIdx.x >> 5;
    int lane = threadIdx.x & 31;
    int row  = blockIdx.x * 8 + wave;       // 0 .. NZ*SEQ-1
    int z    = row >> 12;                   // row / SEQ
    const __bf16* q  = Qb   + (size_t)row * DKH;
    const float*  ks = Ksum + (size_t)z * DKH;
    float s = 0.f;
    #pragma unroll
    for (int i = 0; i < 4; ++i) {
        int idx = lane + 32 * i;
        s += (float)q[idx] * ks[idx];
    }
    #pragma unroll
    for (int off = 16; off > 0; off >>= 1) s += __shfl_xor(s, off, 32);
    if (lane == 0) Zden[row] = 1.0f / (s + 1e-6f);
}

// ---------------------------------------------------------------------------
// Mode-specific epilogue store for one element of the D tile.
// ---------------------------------------------------------------------------
template <int MODE>
__device__ __forceinline__ void store_elem(void* __restrict__ Cv,
                                           const float* __restrict__ Zden,
                                           int z, int m, int c, float val) {
    if constexpr (MODE == MODE_OUT) {
        float* C = (float*)Cv;
        C[(size_t)m * D_MODEL + c] = val;
    } else if constexpr (MODE == MODE_Q) {
        __bf16* C = (__bf16*)Cv;
        int bb = m >> 12, pos = m & (SEQ - 1);
        int h = c >> 7, e = c & (DKH - 1);
        float v = fmaxf(val, 0.f) * __expf(-(float)pos * (1.0f / SEQ));
        C[(((size_t)(bb * NHEADS + h) * SEQ + pos) << 7) + e] = (__bf16)v;
    } else if constexpr (MODE == MODE_KT) {
        __bf16* C = (__bf16*)Cv;
        int bb = m >> 12, pos = m & (SEQ - 1);
        int h = c >> 7, e = c & (DKH - 1);
        float v = fmaxf(val, 0.f) * __expf(-(float)pos * (1.0f / SEQ));
        C[(((size_t)(bb * NHEADS + h) * DKH + e) << 12) + pos] = (__bf16)v;
    } else if constexpr (MODE == MODE_VT) {
        __bf16* C = (__bf16*)Cv;
        int bb = m >> 12, pos = m & (SEQ - 1);
        int h = c >> 7, e = c & (DKH - 1);
        C[(((size_t)(bb * NHEADS + h) * DKH + e) << 12) + pos] = (__bf16)val;
    } else if constexpr (MODE == MODE_KVT) {
        // D[m=d][c=e]  ->  KVT[z][e][d]
        __bf16* C = (__bf16*)Cv + (size_t)z * DKH * DKH;
        C[(size_t)c * DKH + m] = (__bf16)val;
    } else if constexpr (MODE == MODE_ATTN) {
        int bb = z >> 4, h = z & (NHEADS - 1);
        float v = val * Zden[(size_t)z * SEQ + m];
        __bf16* C = (__bf16*)Cv;
        C[(size_t)(bb * SEQ + m) * D_MODEL + h * DKH + c] = (__bf16)v;
    }
}

// ---------------------------------------------------------------------------
// Generic bf16 WMMA GEMM:  D[m][n] = sum_k A[m][k] * Bt[n][k]
// Block: 256 threads = 8 waves (2 along M x 4 along N)
// Wave tile: 64(M) x 32(N) = 4x2 subtiles of 16x16; block tile 128x128.
// USE_LDS: stage the shared 128x32 A slab through LDS via CDNA5
// async global->LDS copies (ASYNCcnt), double-buffered, ds_load_b128 reads.
// ---------------------------------------------------------------------------
template <int MODE, bool USE_LDS>
__global__ __launch_bounds__(256)
void gemm_wmma(const __bf16* __restrict__ A, const __bf16* __restrict__ Bt,
               void* __restrict__ Cv, const float* __restrict__ Zden,
               int lda, int ldb, int K,
               long long strideA, long long strideB) {
    const int tid    = threadIdx.x;
    const int lane   = tid & 31;
    const int wave   = tid >> 5;
    const int waveM  = wave & 1;       // 0..1
    const int waveN  = wave >> 1;      // 0..3
    const int laneLo = lane & 15;
    const int laneHi = lane >> 4;      // 0 or 1

    const __bf16* Ab = A  + (long long)blockIdx.z * strideA;
    const __bf16* Bb = Bt + (long long)blockIdx.z * strideB;

    const int mBase = blockIdx.x * 128 + waveM * 64;
    const int nBase = blockIdx.y * 128 + waveN * 32;

    // B fragment (32x16, 16-bit): lane col = laneLo (row of Bt);
    // lanes 0-15 hold K=0..7,16..23, lanes 16-31 hold K=8..15,24..31.
    const __bf16* bRow[2];
    #pragma unroll
    for (int j = 0; j < 2; ++j)
        bRow[j] = Bb + (size_t)(nBase + j * 16 + laneLo) * ldb + laneHi * 8;

    v8f acc[4][2];
    #pragma unroll
    for (int i = 0; i < 4; ++i)
        #pragma unroll
        for (int j = 0; j < 2; ++j)
            acc[i][j] = (v8f){0.f, 0.f, 0.f, 0.f, 0.f, 0.f, 0.f, 0.f};

    if constexpr (USE_LDS) {
        __shared__ char smem[2 * LDS_BUF];
        const int mBlock = blockIdx.x * 128;

        // Issue async copies of the 128x32 A slab at k-offset kk into
        // LDS buffer at byte offset bufByte. 512 16B chunks / 256 threads.
        auto issueA = [&](int kk, unsigned bufByte) {
            #pragma unroll
            for (int s = 0; s < 2; ++s) {
                int c   = tid + s * 256;
                int row = c >> 2;            // 0..127
                int off = (c & 3) * 16;      // byte offset within 64B k-slab
                const void* g =
                    (const char*)(Ab + (size_t)(mBlock + row) * lda + kk) + off;
                unsigned l = (unsigned)(uintptr_t)(smem + bufByte +
                                                   row * LDS_PITCH + off);
                asm volatile("global_load_async_to_lds_b128 %0, %1, off"
                             :: "v"(l), "v"(g) : "memory");
            }
        };

        // A fragment LDS offsets: row = waveM*64 + i*16 + laneLo,
        // byte chunks at laneHi*16 and +32 (K halves per ISA A layout).
        unsigned aOff[4];
        #pragma unroll
        for (int i = 0; i < 4; ++i)
            aOff[i] = (unsigned)((waveM * 64 + i * 16 + laneLo) * LDS_PITCH +
                                 laneHi * 16);

        issueA(0, 0);
        unsigned bufByte = 0;
        for (int k0 = 0; k0 < K; k0 += 32) {
            asm volatile("s_wait_asynccnt 0x0" ::: "memory");
            __syncthreads();                       // slab visible to all waves
            if (k0 + 32 < K)
                issueA(k0 + 32, bufByte ^ LDS_BUF);

            FragBF a[4], b[2];
            #pragma unroll
            for (int j = 0; j < 2; ++j) {
                b[j].h[0] = *(const v8bf*)(bRow[j] + k0);
                b[j].h[1] = *(const v8bf*)(bRow[j] + k0 + 16);
            }
            __builtin_prefetch((const void*)(bRow[0] + k0 + 32), 0, 1);
            #pragma unroll
            for (int i = 0; i < 4; ++i) {
                const char* p = smem + bufByte + aOff[i];
                a[i].h[0] = *(const v8bf*)(p);
                a[i].h[1] = *(const v8bf*)(p + 32);
            }
            // Pin all 8 ds_load_b128 + 4 global_load_b128 before the MMA
            // block so dependency waits are staggered, not load->wait0->mma.
            SCHED_FENCE();

            #pragma unroll
            for (int i = 0; i < 4; ++i)
                #pragma unroll
                for (int j = 0; j < 2; ++j)
                    acc[i][j] = __builtin_amdgcn_wmma_f32_16x16x32_bf16(
                        false, a[i].v, false, b[j].v,
                        (short)0, acc[i][j], false, false);
            bufByte ^= LDS_BUF;
        }
    } else {
        // Direct-from-global fragments (small-K / latency-sensitive stages).
        const __bf16* aRow[4];
        #pragma unroll
        for (int i = 0; i < 4; ++i)
            aRow[i] = Ab + (size_t)(mBase + i * 16 + laneLo) * lda + laneHi * 8;

        for (int k0 = 0; k0 < K; k0 += 32) {
            FragBF a[4], b[2];
            #pragma unroll
            for (int j = 0; j < 2; ++j) {
                b[j].h[0] = *(const v8bf*)(bRow[j] + k0);
                b[j].h[1] = *(const v8bf*)(bRow[j] + k0 + 16);
            }
            #pragma unroll
            for (int i = 0; i < 4; ++i) {
                a[i].h[0] = *(const v8bf*)(aRow[i] + k0);
                a[i].h[1] = *(const v8bf*)(aRow[i] + k0 + 16);
            }
            __builtin_prefetch((const void*)(aRow[0] + k0 + 32), 0, 1);
            __builtin_prefetch((const void*)(bRow[0] + k0 + 32), 0, 1);
            SCHED_FENCE();

            #pragma unroll
            for (int i = 0; i < 4; ++i)
                #pragma unroll
                for (int j = 0; j < 2; ++j)
                    acc[i][j] = __builtin_amdgcn_wmma_f32_16x16x32_bf16(
                        false, a[i].v, false, b[j].v,
                        (short)0, acc[i][j], false, false);
        }
    }

    // Epilogue. C/D layout: VGPR r, lanes 0-15 -> M=r, lanes 16-31 -> M=r+8,
    // N = lane % 16.
    #pragma unroll
    for (int i = 0; i < 4; ++i) {
        #pragma unroll
        for (int j = 0; j < 2; ++j) {
            #pragma unroll
            for (int r = 0; r < 8; ++r) {
                float val = acc[i][j][r];
                int m = mBase + i * 16 + r + laneHi * 8;
                int c = nBase + j * 16 + laneLo;
                store_elem<MODE>(Cv, Zden, blockIdx.z, m, c, val);
            }
        }
    }
}

// ---------------------------------------------------------------------------
// Host-side orchestration
// ---------------------------------------------------------------------------
extern "C" void kernel_launch(void* const* d_in, const int* in_sizes, int n_in,
                              void* d_out, int out_size, void* d_ws, size_t ws_size,
                              hipStream_t stream) {
    const float* x  = (const float*)d_in[0];
    const float* Wq = (const float*)d_in[1];
    const float* Wk = (const float*)d_in[3];
    const float* Wv = (const float*)d_in[5];
    const float* Wo = (const float*)d_in[7];
    float* out = (float*)d_out;

    // Workspace layout (bytes)
    char* ws = (char*)d_ws;
    size_t o = 0;
    __bf16* xb   = (__bf16*)(ws + o); o += (size_t)MTOT * D_MODEL * 2;     // 64 MB
    __bf16* wqb  = (__bf16*)(ws + o); o += (size_t)D_MODEL * D_MODEL * 2;  // 8 MB
    __bf16* wkb  = (__bf16*)(ws + o); o += (size_t)D_MODEL * D_MODEL * 2;
    __bf16* wvb  = (__bf16*)(ws + o); o += (size_t)D_MODEL * D_MODEL * 2;
    __bf16* wob  = (__bf16*)(ws + o); o += (size_t)D_MODEL * D_MODEL * 2;
    __bf16* Qb   = (__bf16*)(ws + o); o += (size_t)NZ * SEQ * DKH * 2;     // 64 MB
    __bf16* KTb  = (__bf16*)(ws + o); o += (size_t)NZ * DKH * SEQ * 2;     // 64 MB
    __bf16* VTb  = (__bf16*)(ws + o); o += (size_t)NZ * DKH * SEQ * 2;     // 64 MB
    __bf16* KVTb = (__bf16*)(ws + o); o += (size_t)NZ * DKH * DKH * 2;     // 2 MB
    float*  Ksum = (float*)(ws + o);  o += (size_t)NZ * DKH * 4;
    float*  Zden = (float*)(ws + o);  o += (size_t)NZ * SEQ * 4;
    __bf16* attn = (__bf16*)(ws + o); o += (size_t)MTOT * D_MODEL * 2;     // 64 MB
    (void)ws_size; (void)in_sizes; (void)n_in; (void)out_size;

    // 1) convert activations + weights to bf16
    {
        int nx = MTOT * D_MODEL;          // 33,554,432
        int nw = D_MODEL * D_MODEL;       // 4,194,304
        cvt_f32_to_bf16<<<nx / 1024, 256, 0, stream>>>(x,  xb);
        cvt_f32_to_bf16<<<nw / 1024, 256, 0, stream>>>(Wq, wqb);
        cvt_f32_to_bf16<<<nw / 1024, 256, 0, stream>>>(Wk, wkb);
        cvt_f32_to_bf16<<<nw / 1024, 256, 0, stream>>>(Wv, wvb);
        cvt_f32_to_bf16<<<nw / 1024, 256, 0, stream>>>(Wo, wob);
    }

    dim3 blk(256);
    dim3 gProj(MTOT / 128, D_MODEL / 128, 1);   // 128 x 16

    // 2) Q/K/V projections (fused relu + decay + head split / transpose)
    gemm_wmma<MODE_Q , true><<<gProj, blk, 0, stream>>>(
        xb, wqb, (void*)Qb, nullptr, D_MODEL, D_MODEL, D_MODEL, 0, 0);
    gemm_wmma<MODE_KT, true><<<gProj, blk, 0, stream>>>(
        xb, wkb, (void*)KTb, nullptr, D_MODEL, D_MODEL, D_MODEL, 0, 0);
    gemm_wmma<MODE_VT, true><<<gProj, blk, 0, stream>>>(
        xb, wvb, (void*)VTb, nullptr, D_MODEL, D_MODEL, D_MODEL, 0, 0);

    // 3) Ksum over sequence
    ksum_kernel<<<(NZ * DKH) / 8, blk, 0, stream>>>(KTb, Ksum);

    // 4) KV state per head: KV = K^T @ V  (M=128, N=128, K=4096), stored transposed
    gemm_wmma<MODE_KVT, true><<<dim3(1, 1, NZ), blk, 0, stream>>>(
        KTb, VTb, (void*)KVTb, nullptr,
        SEQ, SEQ, SEQ,
        (long long)DKH * SEQ, (long long)DKH * SEQ);

    // 5) Z = 1 / (Q . Ksum + eps)
    zden_kernel<<<(NZ * SEQ) / 8, blk, 0, stream>>>(Qb, Ksum, Zden);

    // 6) out = (Q @ KV) * Z, repacked to [b, n, h*d]  (K=128: direct path)
    gemm_wmma<MODE_ATTN, false><<<dim3(SEQ / 128, 1, NZ), blk, 0, stream>>>(
        Qb, KVTb, (void*)attn, Zden,
        DKH, DKH, DKH,
        (long long)SEQ * DKH, (long long)DKH * DKH);

    // 7) final projection with Wo -> f32 output
    gemm_wmma<MODE_OUT, true><<<gProj, blk, 0, stream>>>(
        attn, wob, (void*)out, nullptr, D_MODEL, D_MODEL, D_MODEL, 0, 0);
}
